// AudioEncoder_14757507629895
// MI455X (gfx1250) — matrix operand order repaired and checked by reading, not compiled, not run on hardware
//
#include <hip/hip_runtime.h>

// ---------------------------------------------------------------------------
// Types for WMMA bf16 16x16x32 (gfx1250, wave32)
// ---------------------------------------------------------------------------
typedef __attribute__((ext_vector_type(16))) __bf16 v16bf;
typedef __attribute__((ext_vector_type(8)))  float  v8f;

struct alignas(16) Q16 { unsigned int x, y, z, w; };
union Frag { v16bf v; Q16 q[2]; };

__device__ __forceinline__ unsigned short f2bf(float f) {
    unsigned int u = __float_as_uint(f);
    u += 0x7FFFu + ((u >> 16) & 1u);   // round-to-nearest-even
    return (unsigned short)(u >> 16);
}

// ---------------------------------------------------------------------------
// fp32 -> bf16 conversion helpers
// ---------------------------------------------------------------------------
__global__ void k_f32_to_bf16(const float* __restrict__ in,
                              unsigned short* __restrict__ out, long n) {
    long i = (long)blockIdx.x * blockDim.x + threadIdx.x;
    if (i < n) out[i] = f2bf(in[i]);
}

// in: [C][Tin] fp32 ; out: [Tout][C] bf16, rows t>=Tin zeroed
__global__ void k_transpose_pad_bf16(const float* __restrict__ in,
                                     unsigned short* __restrict__ out,
                                     int C, int Tin, int Tout) {
    long i = (long)blockIdx.x * blockDim.x + threadIdx.x;
    long total = (long)Tout * C;
    if (i >= total) return;
    int t = (int)(i / C);
    int c = (int)(i % C);
    out[i] = (t < Tin) ? f2bf(in[(long)c * Tin + t]) : (unsigned short)0;
}

// in: [R][Kin] fp32 ; out: [R][Kout] bf16, cols k>=Kin zeroed
__global__ void k_pad_rows_bf16(const float* __restrict__ in,
                                unsigned short* __restrict__ out,
                                int R, int Kin, int Kout) {
    long i = (long)blockIdx.x * blockDim.x + threadIdx.x;
    long total = (long)R * Kout;
    if (i >= total) return;
    int r = (int)(i / Kout);
    int k = (int)(i % Kout);
    out[i] = (k < Kin) ? f2bf(in[(long)r * Kin + k]) : (unsigned short)0;
}

// ---------------------------------------------------------------------------
// bf16 WMMA GEMM: C[M,N] = A[M,K] * Bt[N,K]^T (+ bias[col])
// A row-major [M][K], Bt row-major [Nrows][K] (B transposed), K % 32 == 0,
// M % 128 == 0. C stored at C[row*ldcr + col*ldcc] (fp32).
//
// Block = 256 threads = 8 waves (2 along M x 4 along N), tile 128x128, BK=32.
// Each wave computes 64x32 = 4x2 v_wmma_f32_16x16x32_bf16 per K-step.
// Global->LDS staging uses the CDNA5 async-LDS path
// (global_load_async_to_lds_b128, ASYNCcnt) with double-buffered LDS:
// 4 async ops per wave per tile; in-order completion => s_wait_asynccnt 0x4
// guarantees the previous tile has landed while the next streams in.
// ---------------------------------------------------------------------------
#define GBM 128
#define GBN 128
#define GBK 32
#define LDP 40   // padded LDS row pitch in halfs (32 data + 8 pad)

__global__ __launch_bounds__(256)
void k_wmma_gemm_bf16(const unsigned short* __restrict__ A,
                      const unsigned short* __restrict__ Bt,
                      float* __restrict__ C,
                      const float* __restrict__ bias,
                      int M, int Nrows, int K,
                      long ldcr, long ldcc) {
    __shared__ unsigned short sA[2][GBM * LDP];
    __shared__ unsigned short sB[2][GBN * LDP];

    const int tid  = threadIdx.x;
    const int lane = tid & 31;
    const int wid  = tid >> 5;
    const int wm   = wid & 1;     // 2 waves along M (64 rows each)
    const int wn   = wid >> 1;    // 4 waves along N (32 cols each)
    const int m0   = blockIdx.x * GBM;
    const int n0   = blockIdx.y * GBN;

    const int mrow = lane & 15;
    const int hi   = lane >> 4;

    v8f acc[4][2] = {};

    // issue one 128x32 A tile + 128x32 B tile into LDS buffer p (async)
    auto issue = [&](int k0, int p) {
#pragma unroll
        for (int i = 0; i < 2; ++i) {
            int idx = tid + i * 256;          // 512 x 16B chunks
            int r = idx >> 2, c = idx & 3;
            const unsigned short* gp = A + (long)(m0 + r) * K + k0 + c * 8;
            unsigned dst = (unsigned)(uintptr_t)(const void*)(&sA[p][r * LDP + c * 8]);
            asm volatile("global_load_async_to_lds_b128 %0, %1, off"
                         :: "v"(dst), "v"((unsigned long long)(uintptr_t)gp)
                         : "memory");
        }
#pragma unroll
        for (int i = 0; i < 2; ++i) {
            int idx = tid + i * 256;
            int r = idx >> 2, c = idx & 3;
            unsigned short* lp = &sB[p][r * LDP + c * 8];
            if (n0 + r < Nrows) {
                const unsigned short* gp = Bt + (long)(n0 + r) * K + k0 + c * 8;
                unsigned dst = (unsigned)(uintptr_t)(void*)lp;
                asm volatile("global_load_async_to_lds_b128 %0, %1, off"
                             :: "v"(dst), "v"((unsigned long long)(uintptr_t)gp)
                             : "memory");
            } else {
                Q16 z; z.x = z.y = z.z = z.w = 0u;
                *(Q16*)lp = z;                // zero-pad OOB columns
            }
        }
    };

    const int ktiles = K / GBK;
    issue(0, 0);
    for (int kt = 0; kt < ktiles; ++kt) {
        const int p = kt & 1;
        if (kt + 1 < ktiles) {
            issue((kt + 1) * GBK, p ^ 1);
            asm volatile("s_wait_asynccnt 0x4" ::: "memory");
        } else {
            asm volatile("s_wait_asynccnt 0x0" ::: "memory");
        }
        __syncthreads();

        // fragments per documented gfx1250 VGPR layouts
        Frag fa[4], fb[2];
#pragma unroll
        for (int f = 0; f < 4; ++f) {
            int arow = wm * 64 + f * 16 + mrow;
            // A: lanes<16 hold K {0..7,16..23}; lanes>=16 hold K {8..15,24..31}
            fa[f].q[0] = *(const Q16*)(&sA[p][arow * LDP + hi * 8]);
            fa[f].q[1] = *(const Q16*)(&sA[p][arow * LDP + 16 + hi * 8]);
        }
#pragma unroll
        for (int f = 0; f < 2; ++f) {
            int brow = wn * 32 + f * 16 + mrow;
            // B (as [N][K]): lanes<16 hold K {0..7,16..23}; lanes>=16 hold K {8..15,24..31}
            fb[f].q[0] = *(const Q16*)(&sB[p][brow * LDP + hi * 8]);
            fb[f].q[1] = *(const Q16*)(&sB[p][brow * LDP + 16 + hi * 8]);
        }
#pragma unroll
        for (int i = 0; i < 4; ++i)
#pragma unroll
            for (int j = 0; j < 2; ++j)
                acc[i][j] = __builtin_amdgcn_wmma_f32_16x16x32_bf16(
                    false, fa[i].v, false, fb[j].v,
                    (short)0, acc[i][j], false, false);
        __syncthreads();
    }

    // store: C/D layout: VGPR r -> M=r (lanes 0-15) / M=r+8 (lanes 16-31)
#pragma unroll
    for (int i = 0; i < 4; ++i)
#pragma unroll
        for (int j = 0; j < 2; ++j)
#pragma unroll
            for (int r = 0; r < 8; ++r) {
                int row = m0 + wm * 64 + i * 16 + r + hi * 8;
                int col = n0 + wn * 32 + j * 16 + mrow;
                if (col < Nrows) {
                    float v = acc[i][j][r];
                    if (bias) v += bias[col];
                    C[(long)row * ldcr + (long)col * ldcc] = v;
                }
            }
}

// ---------------------------------------------------------------------------
// Device-scope sense barrier for the persistent recurrence kernel
// ---------------------------------------------------------------------------
__device__ __forceinline__ void gridBarrier(unsigned int* bar, unsigned int nb) {
    __syncthreads();
    if (threadIdx.x == 0) {
        __builtin_amdgcn_fence(__ATOMIC_RELEASE, "agent");
        unsigned int gen = __hip_atomic_load(&bar[1], __ATOMIC_RELAXED,
                                             __HIP_MEMORY_SCOPE_AGENT);
        unsigned int arrived = __hip_atomic_fetch_add(&bar[0], 1u, __ATOMIC_ACQ_REL,
                                                      __HIP_MEMORY_SCOPE_AGENT);
        if (arrived + 1u == nb) {
            __hip_atomic_store(&bar[0], 0u, __ATOMIC_RELAXED,
                               __HIP_MEMORY_SCOPE_AGENT);
            __hip_atomic_fetch_add(&bar[1], 1u, __ATOMIC_RELEASE,
                                   __HIP_MEMORY_SCOPE_AGENT);
        } else {
            while (__hip_atomic_load(&bar[1], __ATOMIC_RELAXED,
                                     __HIP_MEMORY_SCOPE_AGENT) == gen) {
                __builtin_amdgcn_s_sleep(1);
            }
        }
        __builtin_amdgcn_fence(__ATOMIC_ACQUIRE, "agent");
    }
    __syncthreads();
}

// ---------------------------------------------------------------------------
// Persistent LSTM recurrence (one layer, both "directions" run forward).
// X:      [T][8192] precomputed input gate contributions (d*4096+gate*1024+j)
// Whh_bf: [2][4096][1024] bf16 (L2-resident: 16.8 MB -> half the L2 stream
//         of fp32; this loop IS the runtime bottleneck: ~860 sequential steps)
// hbuf:   [pingpong 2][dir 2][1024] fp32, zeroed before launch
// Yout:   mode 0 -> bf16 [T][2048]   (layer-0 concat output, feeds X1 GEMM)
//         mode 1 -> bf16 [2048][Tp]  (pre-transposed for projection GEMM)
// Grid: 128 blocks x 256 threads; block b owns 16 hidden units of
// direction d = b/64 -> c state lives in registers of threads 0..15.
// ---------------------------------------------------------------------------
#define RB_GRID 128

__global__ __launch_bounds__(256)
void k_lstm_recur(const float* __restrict__ X,
                  const unsigned short* __restrict__ Whh_bf,
                  const float* __restrict__ b_ih,
                  const float* __restrict__ b_hh,
                  float* __restrict__ hbuf,
                  unsigned short* __restrict__ Yout,
                  int T, int Tp, int mode,
                  unsigned int* __restrict__ bar) {
    const int tid  = threadIdx.x;
    const int lane = tid & 31;
    const int wid  = tid >> 5;            // 0..7
    const int blk  = blockIdx.x;
    const int d    = blk >> 6;            // direction 0/1
    const int u0   = (blk & 63) * 16;     // first hidden unit (within direction)

    __shared__ float h_lds[1024];
    __shared__ float gates[64];           // [gate*16 + ui] preactivations

    float c_state = 0.0f;                 // threads 0..15 own unit u0+tid

    for (int t = 0; t < T; ++t) {
        // stage h_{t-1} of this direction into LDS
        const float* hsrc = hbuf + (((t & 1) * 2 + d) << 10);
        for (int k = tid; k < 1024; k += 256) h_lds[k] = hsrc[k];
        __syncthreads();

        // 64 gate rows per block, 8 per wave: 1024-long dot each,
        // weights streamed as packed bf16 pairs (1 u32 load -> 2 fp32 FMAs)
#pragma unroll
        for (int rr = 0; rr < 8; ++rr) {
            int row  = wid * 8 + rr;              // 0..63
            int gate = row >> 4;                  // i,f,g,o
            int ui   = row & 15;
            int grow = gate * 1024 + u0 + ui;     // 0..4095 within direction
            const unsigned int* w32 = (const unsigned int*)
                (Whh_bf + ((long)d * 4096 + grow) * 1024);
            float sum = 0.f;
#pragma unroll 4
            for (int kp = lane; kp < 512; kp += 32) {
                unsigned int pw = w32[kp];
                float2 hv = *(const float2*)(h_lds + 2 * kp);
                sum = fmaf(__uint_as_float(pw << 16),         hv.x, sum);
                sum = fmaf(__uint_as_float(pw & 0xFFFF0000u), hv.y, sum);
            }
#pragma unroll
            for (int off = 16; off; off >>= 1)
                sum += __shfl_xor(sum, off, 32);
            if (lane == 0) {
                int gi = d * 4096 + grow;
                gates[gate * 16 + ui] =
                    sum + X[(long)t * 8192 + gi] + b_ih[gi] + b_hh[gi];
            }
        }
        __syncthreads();

        // c/h update for the 16 owned units
        if (tid < 16) {
            float gi_ = gates[0 * 16 + tid];
            float gf_ = gates[1 * 16 + tid];
            float gg_ = gates[2 * 16 + tid];
            float go_ = gates[3 * 16 + tid];
            float si = 1.f / (1.f + __expf(-gi_));
            float sf = 1.f / (1.f + __expf(-gf_));
            float so = 1.f / (1.f + __expf(-go_));
            c_state  = sf * c_state + si * tanhf(gg_);
            float h  = so * tanhf(c_state);
            int j = u0 + tid;
            hbuf[((((t + 1) & 1) * 2 + d) << 10) + j] = h;  // ping-pong write
            unsigned short hb = f2bf(h);
            int u = d * 1024 + j;                           // concat index
            if (mode == 0) Yout[(long)t * 2048 + u] = hb;
            else           Yout[(long)u * Tp + t]   = hb;
        }

        gridBarrier(bar, RB_GRID);
    }
}

// ---------------------------------------------------------------------------
// Host side
// ---------------------------------------------------------------------------
extern "C" void kernel_launch(void* const* d_in, const int* in_sizes, int n_in,
                              void* d_out, int out_size, void* d_ws, size_t ws_size,
                              hipStream_t stream) {
    const float* spec  = (const float*)d_in[0];   // [128][860]
    const float* Wih0  = (const float*)d_in[1];   // [2][4096][128]
    const float* Whh0  = (const float*)d_in[2];   // [2][4096][1024]
    const float* bih0  = (const float*)d_in[3];   // [2][4096]
    const float* bhh0  = (const float*)d_in[4];
    const float* Wih1  = (const float*)d_in[5];   // [2][4096][2048]
    const float* Whh1  = (const float*)d_in[6];
    const float* bih1  = (const float*)d_in[7];
    const float* bhh1  = (const float*)d_in[8];
    const float* Wproj = (const float*)d_in[9];   // [240][860]
    const float* bproj = (const float*)d_in[10];  // [240]
    float* out = (float*)d_out;                   // [2048][240]

    const int T  = 860;
    const int Tp = 864;                           // pad to multiple of 32

    char* ws = (char*)d_ws;
    size_t off = 0;
    auto alloc = [&](size_t bytes) -> void* {
        off = (off + 255) & ~(size_t)255;
        void* p = ws + off;
        off += bytes;
        return p;
    };

    unsigned int*   bar     = (unsigned int*)  alloc(256);
    float*          hbuf    = (float*)         alloc(4 * 1024 * sizeof(float));
    unsigned short* Wih0_bf = (unsigned short*)alloc(8192L * 128 * 2);
    unsigned short* specT   = (unsigned short*)alloc((long)Tp * 128 * 2);
    unsigned short* Wih1_bf = (unsigned short*)alloc(8192L * 2048 * 2);
    unsigned short* Whh0_bf = (unsigned short*)alloc(8192L * 1024 * 2);
    unsigned short* Whh1_bf = (unsigned short*)alloc(8192L * 1024 * 2);
    unsigned short* Wp_bf   = (unsigned short*)alloc(240L * Tp * 2);
    unsigned short* Y0_bf   = (unsigned short*)alloc((long)Tp * 2048 * 2);
    unsigned short* resT    = (unsigned short*)alloc(2048L * Tp * 2);
    float*          X       = (float*)         alloc((long)Tp * 8192 * 4); // X0 & X1

    // zero init: barrier + h state + resT (K-pad cols must be 0 for proj GEMM)
    hipMemsetAsync(bar,  0, 256, stream);
    hipMemsetAsync(hbuf, 0, 4 * 1024 * sizeof(float), stream);
    hipMemsetAsync(resT, 0, 2048L * Tp * 2, stream);

    // ---- fp32 -> bf16 conversions / transposes ----
    {
        long n = 8192L * 128;
        k_f32_to_bf16<<<dim3((unsigned)((n + 255) / 256)), 256, 0, stream>>>(
            Wih0, Wih0_bf, n);
    }
    {
        long n = (long)Tp * 128;
        k_transpose_pad_bf16<<<dim3((unsigned)((n + 255) / 256)), 256, 0, stream>>>(
            spec, specT, 128, T, Tp);
    }
    {
        long n = 8192L * 2048;
        k_f32_to_bf16<<<dim3((unsigned)((n + 255) / 256)), 256, 0, stream>>>(
            Wih1, Wih1_bf, n);
    }
    {
        long n = 8192L * 1024;
        k_f32_to_bf16<<<dim3((unsigned)((n + 255) / 256)), 256, 0, stream>>>(
            Whh0, Whh0_bf, n);
        k_f32_to_bf16<<<dim3((unsigned)((n + 255) / 256)), 256, 0, stream>>>(
            Whh1, Whh1_bf, n);
    }
    {
        long n = 240L * Tp;
        k_pad_rows_bf16<<<dim3((unsigned)((n + 255) / 256)), 256, 0, stream>>>(
            Wproj, Wp_bf, 240, T, Tp);
    }

    // ---- X0 = W_ih0 * spec  -> stored transposed as [t][8192] ----
    {
        dim3 g(8192 / GBM, (Tp + GBN - 1) / GBN);
        k_wmma_gemm_bf16<<<g, 256, 0, stream>>>(
            Wih0_bf, specT, X, nullptr, 8192, Tp, 128, 1L, 8192L);
    }

    // ---- layer-0 recurrence -> Y0 bf16 [t][2048] ----
    k_lstm_recur<<<RB_GRID, 256, 0, stream>>>(
        X, Whh0_bf, bih0, bhh0, hbuf, Y0_bf, T, Tp, 0, bar);

    // ---- X1 = W_ih1 * Y0^T -> stored transposed as [t][8192] ----
    {
        dim3 g(8192 / GBM, (Tp + GBN - 1) / GBN);
        k_wmma_gemm_bf16<<<g, 256, 0, stream>>>(
            Wih1_bf, Y0_bf, X, nullptr, 8192, Tp, 2048, 1L, 8192L);
    }

    // ---- layer-1 recurrence -> resT bf16 [2048][Tp] ----
    hipMemsetAsync(bar,  0, 256, stream);
    hipMemsetAsync(hbuf, 0, 4 * 1024 * sizeof(float), stream);
    k_lstm_recur<<<RB_GRID, 256, 0, stream>>>(
        X, Whh1_bf, bih1, bhh1, hbuf, resT, T, Tp, 1, bar);

    // ---- out[2048][240] = resT[2048][Tp] * Wp_bf[240][Tp]^T + b_proj ----
    {
        dim3 g(2048 / GBM, (240 + GBN - 1) / GBN);
        k_wmma_gemm_bf16<<<g, 256, 0, stream>>>(
            resT, Wp_bf, out, bproj, 2048, 240, Tp, 240L, 1L);
    }
}
